// DecoderAttModule_58737972740623
// MI455X (gfx1250) — matrix operand order repaired and checked by reading, not compiled, not run on hardware
//
#include <hip/hip_runtime.h>
#include <hip/hip_bf16.h>
#include <math.h>

// ---------------------------------------------------------------------------
// Problem constants (from the reference)
// ---------------------------------------------------------------------------
#define BB   256     // batch
#define RR   36      // regions
#define FF   2048    // feature dim
#define AA   512     // attention dim
#define EE   1024    // embedding dim
#define DD   1024    // hidden dim
#define VV   10000   // vocab
#define LL   20      // sequence length
#define TT   19      // scan steps (L-1)
#define KTD  (DD + FF + EE + DD)   // 5120 : [h2|favg|e_t|h1]
#define KLG  (FF + DD + DD)        // 4096 : [aw|h1|h2]
#define NG   (4 * DD)              // 4096 gates

typedef __bf16  bf16_t;
typedef __attribute__((ext_vector_type(16))) __bf16          v16bf;
typedef __attribute__((ext_vector_type(8)))  float           v8f;
typedef __attribute__((ext_vector_type(16))) unsigned short  v16u;

// Native f32 -> bf16 (RNE) — lowers to a single v_cvt on gfx1250
__device__ __forceinline__ unsigned short f2bf(float f) {
  return __builtin_bit_cast(unsigned short, static_cast<__bf16>(f));
}

__device__ __forceinline__ float sigf(float x) { return 1.0f / (1.0f + expf(-x)); }

// ---------------------------------------------------------------------------
// Stable descending sort of lens (B=256), one block of 256 threads.
// rank(b) = #{j : len_j > len_b} + #{j<b : len_j == len_b}  (matches stable
// jnp.argsort(-lens)).
// ---------------------------------------------------------------------------
__global__ void k_sort(const int* __restrict__ sizes, int* __restrict__ order,
                       int* __restrict__ lens_sorted) {
  __shared__ int sl[BB];
  int b = threadIdx.x;
  sl[b] = sizes[b];
  __syncthreads();
  int len = sl[b], rank = 0;
  for (int j = 0; j < BB; ++j) {
    int lj = sl[j];
    if (lj > len || (lj == len && j < b)) ++rank;
  }
  order[rank] = b;
  lens_sorted[rank] = len;
}

// ---------------------------------------------------------------------------
// Zero a contiguous 4-byte-multiple region (graph-capture-safe memset)
// ---------------------------------------------------------------------------
__global__ void k_zero_u32(unsigned int* __restrict__ p, size_t n) {
  for (size_t i = blockIdx.x * (size_t)blockDim.x + threadIdx.x; i < n;
       i += (size_t)gridDim.x * blockDim.x) p[i] = 0u;
}

// ---------------------------------------------------------------------------
// f32 -> bf16 convert, and [A|B] row-concat convert (fuses wih/whh along K)
// ---------------------------------------------------------------------------
__global__ void k_convert(const float* __restrict__ src,
                          unsigned short* __restrict__ dst, size_t n) {
  for (size_t i = blockIdx.x * (size_t)blockDim.x + threadIdx.x; i < n;
       i += (size_t)gridDim.x * blockDim.x) dst[i] = f2bf(src[i]);
}

__global__ void k_concat2(const float* __restrict__ A, int Ka,
                          const float* __restrict__ B, int Kb,
                          unsigned short* __restrict__ dst, size_t total) {
  int Kt = Ka + Kb;
  for (size_t i = blockIdx.x * (size_t)blockDim.x + threadIdx.x; i < total;
       i += (size_t)gridDim.x * blockDim.x) {
    size_t nrow = i / Kt;
    int k = (int)(i % Kt);
    float v = (k < Ka) ? A[nrow * Ka + k] : B[nrow * Kb + (k - Ka)];
    dst[i] = f2bf(v);
  }
}

// ---------------------------------------------------------------------------
// favg[b][f] = mean_r feats[order[b]][r][f]
// ---------------------------------------------------------------------------
__global__ void k_favg(const float* __restrict__ feats,
                       const int* __restrict__ order,
                       float* __restrict__ favg) {
  int b = blockIdx.x;
  const float* src = feats + (size_t)order[b] * RR * FF;
  for (int f = threadIdx.x; f < FF; f += blockDim.x) {
    float s = 0.f;
    for (int r = 0; r < RR; ++r) s += src[(size_t)r * FF + f];
    favg[(size_t)b * FF + f] = s * (1.0f / RR);
  }
}

// ---------------------------------------------------------------------------
// bf16 WMMA GEMM: C[M,N'] = A[M,K] * W[N',K]^T + bias (*mask)
//   4 waves / 128-thread block; each wave owns a 16(M) x (NT*16)(N) tile.
//   A fragment loaded once per K-chunk, reused across NT B fragments.
//   Template NT removes all inner-loop branches (full tiles only; the host
//   launches a separate NT<4 kernel for the column remainder, offset n_origin).
//   A/B fragments use contiguous k order (k = 16*g + s); EXEC stays all-ones per wave.
// ---------------------------------------------------------------------------
template <int NT, bool MASK>
__global__ __launch_bounds__(128)
void k_gemm_bf16(const unsigned short* __restrict__ Abf,
                 const unsigned short* __restrict__ Wbf,
                 const float* __restrict__ bias,
                 float* __restrict__ C,
                 int K, int ldc, int n_origin,
                 const int* __restrict__ lens, int t) {
  const int lane  = threadIdx.x & 31;
  const int wave  = threadIdx.x >> 5;
  const int nwav  = blockDim.x >> 5;
  const int m     = lane & 15;     // row in tile (A) / col in tile (B,D)
  const int g     = lane >> 4;     // lane group
  const int mbase = blockIdx.y << 4;
  const int nbase = n_origin + (blockIdx.x * nwav + wave) * (NT * 16);

  const unsigned short* arow  = Abf + (size_t)(mbase + m) * K;
  const unsigned short* wrow0 = Wbf + (size_t)(nbase + m) * K;

  v8f acc[NT];
#pragma unroll
  for (int j = 0; j < NT; ++j) acc[j] = v8f{};

#pragma unroll 2
  for (int k0 = 0; k0 < K; k0 += 32) {
    // A fragment: 16x32 bf16 (lane-group K split 0-15 | 16-31)
    v16u au;
#pragma unroll
    for (int i = 0; i < 8; ++i) {
      au[i]     = arow[k0 + g * 16 + i];
      au[8 + i] = arow[k0 + g * 16 + 8 + i];
    }
    v16bf a = __builtin_bit_cast(v16bf, au);
    // All NT B fragments (load clauses), then NT back-to-back WMMAs
    v16u bu[NT];
#pragma unroll
    for (int j = 0; j < NT; ++j) {
      const unsigned short* wr = wrow0 + (size_t)(j * 16) * K + k0 + g * 16;
#pragma unroll
      for (int i = 0; i < 16; ++i) bu[j][i] = wr[i];
    }
#pragma unroll
    for (int j = 0; j < NT; ++j) {
      v16bf bfrag = __builtin_bit_cast(v16bf, bu[j]);
      acc[j] = __builtin_amdgcn_wmma_f32_16x16x32_bf16(
          false, a, false, bfrag, (short)0, acc[j], false, false);
    }
  }

#pragma unroll
  for (int j = 0; j < NT; ++j) {
    int ncol = nbase + j * 16 + m;
    float bb = bias[ncol];
#pragma unroll
    for (int v = 0; v < 8; ++v) {
      int row = mbase + v + 8 * g;             // D layout: M = v + 8*g
      float val = acc[j][v] + bb;
      if (MASK) val = (t < lens[row] - 1) ? val : 0.0f;
      C[(size_t)row * ldc + ncol] = val;
    }
  }
}

// ---------------------------------------------------------------------------
// img_att GEMM: rows are (b,r) of sorted feats, A converted f32->bf16 on the
// fly (avoids a 38MB staging buffer). C[9216,512] = fts_sorted * af_w^T + af_b
// ---------------------------------------------------------------------------
__global__ __launch_bounds__(128)
void k_gemm_imgatt(const float* __restrict__ feats,
                   const int* __restrict__ order,
                   const unsigned short* __restrict__ Wbf,   // af_w bf16 (512x2048)
                   const float* __restrict__ bias,           // af_b
                   float* __restrict__ C) {                  // (B*R) x 512
  const int N = AA, K = FF;
  const int lane  = threadIdx.x & 31;
  const int wave  = threadIdx.x >> 5;
  const int nwav  = blockDim.x >> 5;
  const int m     = lane & 15;
  const int g     = lane >> 4;
  const int mbase = blockIdx.y << 4;
  const int nbase = (blockIdx.x * nwav + wave) << 6;

  const int row = mbase + m;
  const int b = row / RR, r = row % RR;
  const float* arow = feats + ((size_t)order[b] * RR + r) * FF;
  const unsigned short* wrow0 = Wbf + (size_t)(nbase + m) * K;

  v8f acc[4];
#pragma unroll
  for (int j = 0; j < 4; ++j) acc[j] = v8f{};

  for (int k0 = 0; k0 < K; k0 += 32) {
    v16u au;
#pragma unroll
    for (int i = 0; i < 8; ++i) {
      au[i]     = f2bf(arow[k0 + g * 16 + i]);
      au[8 + i] = f2bf(arow[k0 + g * 16 + 8 + i]);
    }
    v16bf a = __builtin_bit_cast(v16bf, au);
    v16u bu[4];
#pragma unroll
    for (int j = 0; j < 4; ++j) {
      const unsigned short* wr = wrow0 + (size_t)(j * 16) * K + k0 + g * 16;
#pragma unroll
      for (int i = 0; i < 16; ++i) bu[j][i] = wr[i];
    }
#pragma unroll
    for (int j = 0; j < 4; ++j) {
      v16bf bfrag = __builtin_bit_cast(v16bf, bu[j]);
      acc[j] = __builtin_amdgcn_wmma_f32_16x16x32_bf16(
          false, a, false, bfrag, (short)0, acc[j], false, false);
    }
  }
#pragma unroll
  for (int j = 0; j < 4; ++j) {
    int ncol = nbase + j * 16 + m;
    float bb = bias[ncol];
#pragma unroll
    for (int v = 0; v < 8; ++v) {
      int orow = mbase + v + 8 * g;
      C[(size_t)orow * N + ncol] = acc[j][v] + bb;
    }
  }
}

// ---------------------------------------------------------------------------
// Build Xtd[b] = [h2 | favg | emb[seq[order[b],t]] | h1]   (bf16, K=5120)
// ---------------------------------------------------------------------------
__global__ void k_build_xtd(const float* __restrict__ h1,
                            const float* __restrict__ h2,
                            const float* __restrict__ favg,
                            const float* __restrict__ emb,
                            const int* __restrict__ seqs,
                            const int* __restrict__ order,
                            int t, unsigned short* __restrict__ Xtd) {
  size_t i = blockIdx.x * (size_t)blockDim.x + threadIdx.x;
  if (i >= (size_t)BB * KTD) return;
  int b = (int)(i / KTD), k = (int)(i % KTD);
  float v;
  if (k < DD)              v = h2[(size_t)b * DD + k];
  else if (k < DD + FF)    v = favg[(size_t)b * FF + (k - DD)];
  else if (k < DD + FF + EE) {
    int tok = seqs[order[b] * LL + t];
    v = emb[(size_t)tok * EE + (k - DD - FF)];
  } else                   v = h1[(size_t)b * DD + (k - DD - FF - EE)];
  Xtd[i] = f2bf(v);
}

// ---------------------------------------------------------------------------
// LSTM pointwise: gates g[b,4D] (i,f,g,o), masked state update, h -> bf16 too
// ---------------------------------------------------------------------------
__global__ void k_lstm_update(const float* __restrict__ gbuf,
                              float* __restrict__ h, float* __restrict__ c,
                              unsigned short* __restrict__ hbf,
                              const int* __restrict__ lens, int t) {
  int i = blockIdx.x * blockDim.x + threadIdx.x;
  if (i >= BB * DD) return;
  int b = i / DD, d = i % DD;
  const float* gr = gbuf + (size_t)b * NG;
  float gi = gr[d], gf = gr[DD + d], gg = gr[2 * DD + d], go = gr[3 * DD + d];
  float c_old = c[i], h_old = h[i];
  float nc = sigf(gf) * c_old + sigf(gi) * tanhf(gg);
  float nh = sigf(go) * tanhf(nc);
  bool act = (t < lens[b] - 1);
  float hv = act ? nh : h_old;
  if (act) { c[i] = nc; h[i] = nh; }
  hbf[i] = f2bf(hv);
}

// ---------------------------------------------------------------------------
// Attention: one block per batch row.
//   att[r]  = at_b + sum_a at_w[a]*relu(img_att[b,r,a] + dec_att[b,a])
//   alpha   = softmax_r(att)
//   aw[b,f] = sum_r alpha[r]*feats[order[b],r,f]  -> bf16 into Xlg[:, 0:F]
// ---------------------------------------------------------------------------
__global__ void k_attention(const float* __restrict__ feats,
                            const int* __restrict__ order,
                            const float* __restrict__ img_att,
                            const float* __restrict__ dec_att,
                            const float* __restrict__ at_w,
                            const float* __restrict__ at_b,
                            unsigned short* __restrict__ Xlg) {
  __shared__ float s_dec[AA];
  __shared__ float s_alpha[RR];
  int b = blockIdx.x;
  for (int a = threadIdx.x; a < AA; a += blockDim.x)
    s_dec[a] = dec_att[(size_t)b * AA + a];
  __syncthreads();

  int wave = threadIdx.x >> 5, lane = threadIdx.x & 31;
  int nwaves = blockDim.x >> 5;
  for (int r = wave; r < RR; r += nwaves) {
    const float* ia = img_att + ((size_t)b * RR + r) * AA;
    float s = 0.f;
    for (int a = lane; a < AA; a += 32)
      s += at_w[a] * fmaxf(ia[a] + s_dec[a], 0.f);
#pragma unroll
    for (int off = 16; off > 0; off >>= 1) s += __shfl_xor(s, off, 32);
    if (lane == 0) s_alpha[r] = s + at_b[0];
  }
  __syncthreads();
  if (threadIdx.x == 0) {
    float mx = -3.4e38f;
    for (int r = 0; r < RR; ++r) mx = fmaxf(mx, s_alpha[r]);
    float sum = 0.f;
    for (int r = 0; r < RR; ++r) { float e = expf(s_alpha[r] - mx); s_alpha[r] = e; sum += e; }
    float inv = 1.0f / sum;
    for (int r = 0; r < RR; ++r) s_alpha[r] *= inv;
  }
  __syncthreads();
  const float* fb = feats + (size_t)order[b] * RR * FF;
  for (int f = threadIdx.x; f < FF; f += blockDim.x) {
    float s = 0.f;
    for (int r = 0; r < RR; ++r) s += s_alpha[r] * fb[(size_t)r * FF + f];
    Xlg[(size_t)b * KLG + f] = f2bf(s);
  }
}

// ---------------------------------------------------------------------------
// Xlg tail: [.., h1 | h2] (bf16 copies already produced by k_lstm_update)
// ---------------------------------------------------------------------------
__global__ void k_build_xlg_tail(const unsigned short* __restrict__ h1bf,
                                 const unsigned short* __restrict__ h2bf,
                                 unsigned short* __restrict__ Xlg) {
  int i = blockIdx.x * blockDim.x + threadIdx.x;
  if (i >= BB * 2 * DD) return;
  int b = i / (2 * DD), k = i % (2 * DD);
  Xlg[(size_t)b * KLG + FF + k] =
      (k < DD) ? h1bf[(size_t)b * DD + k] : h2bf[(size_t)b * DD + (k - DD)];
}

// ---------------------------------------------------------------------------
// Host side
// ---------------------------------------------------------------------------
extern "C" void kernel_launch(void* const* d_in, const int* in_sizes, int n_in,
                              void* d_out, int out_size, void* d_ws, size_t ws_size,
                              hipStream_t stream) {
  (void)in_sizes; (void)n_in; (void)out_size; (void)ws_size;
  const float* feats  = (const float*)d_in[0];
  const int*   seqs   = (const int*)  d_in[1];
  const int*   sizes  = (const int*)  d_in[2];
  const float* emb    = (const float*)d_in[3];
  const float* td_wih = (const float*)d_in[4];
  const float* td_whh = (const float*)d_in[5];
  const float* td_b   = (const float*)d_in[6];
  const float* lg_wih = (const float*)d_in[7];
  const float* lg_whh = (const float*)d_in[8];
  const float* lg_b   = (const float*)d_in[9];
  const float* af_w   = (const float*)d_in[10];
  const float* af_b   = (const float*)d_in[11];
  const float* ad_w   = (const float*)d_in[12];
  const float* ad_b   = (const float*)d_in[13];
  const float* at_w   = (const float*)d_in[14];
  const float* at_b   = (const float*)d_in[15];
  const float* out_w  = (const float*)d_in[16];
  const float* out_b  = (const float*)d_in[17];
  float* out = (float*)d_out;

  // ---- workspace carve-up (256B aligned) ----
  char* ws = (char*)d_ws;
  size_t off = 0;
  auto carve = [&](size_t bytes) -> void* {
    void* p = ws + off;
    off = (off + bytes + 255) & ~(size_t)255;
    return p;
  };
  int*            order    = (int*)carve(BB * sizeof(int));
  int*            lens     = (int*)carve(BB * sizeof(int));
  unsigned short* Wtd_bf   = (unsigned short*)carve((size_t)NG * KTD * 2);   // [td_wih|td_whh]
  unsigned short* Wlg_bf   = (unsigned short*)carve((size_t)NG * KLG * 2);   // [lg_wih|lg_whh]
  unsigned short* outw_bf  = (unsigned short*)carve((size_t)VV * DD * 2);
  unsigned short* afw_bf   = (unsigned short*)carve((size_t)AA * FF * 2);
  unsigned short* adw_bf   = (unsigned short*)carve((size_t)AA * DD * 2);
  float*          favg     = (float*)carve((size_t)BB * FF * 4);
  float*          img_att  = (float*)carve((size_t)BB * RR * AA * 4);
  // contiguous state block (zeroed each call): h1,c1,h2,c2 f32 + h1bf,h2bf
  size_t state_u32 = (size_t)4 * BB * DD + (size_t)BB * DD;
  unsigned int*   state    = (unsigned int*)carve(state_u32 * 4);
  float* h1 = (float*)state;
  float* c1 = h1 + (size_t)BB * DD;
  float* h2 = c1 + (size_t)BB * DD;
  float* c2 = h2 + (size_t)BB * DD;
  unsigned short* h1bf = (unsigned short*)(c2 + (size_t)BB * DD);
  unsigned short* h2bf = h1bf + (size_t)BB * DD;
  unsigned short* Xtd  = (unsigned short*)carve((size_t)BB * KTD * 2);
  unsigned short* Xlg  = (unsigned short*)carve((size_t)BB * KLG * 2);
  float*          gbuf = (float*)carve((size_t)BB * NG * 4);
  float*          dec  = (float*)carve((size_t)BB * AA * 4);

  // ---- one-time (per call) prep ----
  k_sort<<<1, BB, 0, stream>>>(sizes, order, lens);
  k_zero_u32<<<512, 256, 0, stream>>>(state, state_u32);
  k_concat2<<<4096, 256, 0, stream>>>(td_wih, DD + FF + EE, td_whh, DD, Wtd_bf,
                                      (size_t)NG * KTD);
  k_concat2<<<4096, 256, 0, stream>>>(lg_wih, FF + DD, lg_whh, DD, Wlg_bf,
                                      (size_t)NG * KLG);
  k_convert<<<4096, 256, 0, stream>>>(out_w, outw_bf, (size_t)VV * DD);
  k_convert<<<1024, 256, 0, stream>>>(af_w, afw_bf, (size_t)AA * FF);
  k_convert<<<512, 256, 0, stream>>>(ad_w, adw_bf, (size_t)AA * DD);
  k_favg<<<BB, 256, 0, stream>>>(feats, order, favg);
  // img_att: N=512 -> 8 64-wide tiles -> 2 blocks x 4 waves; M=9216 -> 576
  k_gemm_imgatt<<<dim3(2, (BB * RR) / 16), 128, 0, stream>>>(feats, order,
                                                             afw_bf, af_b,
                                                             img_att);

  // grids: x = (#64-wide tiles)/4 blocks of 4 waves, y = M/16
  const dim3 grd_gate(NG / 64 / 4, BB / 16);         // 16 x 16
  const dim3 grd_dec(AA / 64 / 4, BB / 16);          // 2 x 16
  const dim3 grd_voc((VV / 64) / 4, BB / 16);        // 39 x 16 (9984 cols)
  const dim3 grd_voc_rem(1, BB / 16);                // last 16 cols @ 9984

  // ---- recurrent steps ----
  for (int t = 0; t < TT; ++t) {
    k_build_xtd<<<(BB * KTD + 255) / 256, 256, 0, stream>>>(h1, h2, favg, emb,
                                                            seqs, order, t, Xtd);
    k_gemm_bf16<4, false><<<grd_gate, 128, 0, stream>>>(
        Xtd, Wtd_bf, td_b, gbuf, KTD, NG, 0, nullptr, 0);
    k_lstm_update<<<(BB * DD + 255) / 256, 256, 0, stream>>>(gbuf, h1, c1, h1bf,
                                                             lens, t);
    k_gemm_bf16<4, false><<<grd_dec, 128, 0, stream>>>(
        h1bf, adw_bf, ad_b, dec, DD, AA, 0, nullptr, 0);
    k_attention<<<BB, 256, 0, stream>>>(feats, order, img_att, dec, at_w, at_b,
                                        Xlg);
    k_build_xlg_tail<<<(BB * 2 * DD + 255) / 256, 256, 0, stream>>>(h1bf, h2bf,
                                                                    Xlg);
    k_gemm_bf16<4, false><<<grd_gate, 128, 0, stream>>>(
        Xlg, Wlg_bf, lg_b, gbuf, KLG, NG, 0, nullptr, 0);
    k_lstm_update<<<(BB * DD + 255) / 256, 256, 0, stream>>>(gbuf, h2, c2, h2bf,
                                                             lens, t);
    // masked vocab logits written straight into d_out[(b), t, :]
    k_gemm_bf16<4, true><<<grd_voc, 128, 0, stream>>>(
        h2bf, outw_bf, out_b, out + (size_t)t * VV, DD, TT * VV, 0, lens, t);
    k_gemm_bf16<1, true><<<grd_voc_rem, 32, 0, stream>>>(
        h2bf, outw_bf, out_b, out + (size_t)t * VV, DD, TT * VV, (VV / 64) * 64,
        lens, t);
  }
}
